// SelfAttention1d_72808285601884
// MI455X (gfx1250) — matrix operand order repaired and checked by reading, not compiled, not run on hardware
//
#include <hip/hip_runtime.h>
#include <hip/hip_bf16.h>

typedef _Float16 v8h  __attribute__((ext_vector_type(8)));
typedef _Float16 v16h __attribute__((ext_vector_type(16)));
typedef float    v8f  __attribute__((ext_vector_type(8)));
typedef unsigned int u32x4 __attribute__((ext_vector_type(4)));
typedef int          i32x8 __attribute__((ext_vector_type(8)));
typedef int          i32x4 __attribute__((ext_vector_type(4)));

#define B_  4
#define C_  256
#define L_  2048
#define H_  8
#define D_  32
#define G_  32
#define CPG 8              // channels per group
#define AST 264            // padded LDS stride (halves) for A tiles (256 + 8)
#define BST 264            // padded LDS stride (halves) for Bt tiles
#define PST 40             // padded LDS stride (halves) for P tiles (32 + 8)

__device__ __forceinline__ v16h ldfrag(const _Float16* p0, const _Float16* p1) {
    v8h lo = *(const v8h*)p0;
    v8h hi = *(const v8h*)p1;
    return __builtin_shufflevector(lo, hi, 0,1,2,3,4,5,6,7,8,9,10,11,12,13,14,15);
}

__device__ __forceinline__ float redmax16(float v) {
    v = fmaxf(v, __shfl_xor(v, 1));
    v = fmaxf(v, __shfl_xor(v, 2));
    v = fmaxf(v, __shfl_xor(v, 4));
    v = fmaxf(v, __shfl_xor(v, 8));
    return v;
}
__device__ __forceinline__ float redsum16(float v) {
    v += __shfl_xor(v, 1);
    v += __shfl_xor(v, 2);
    v += __shfl_xor(v, 4);
    v += __shfl_xor(v, 8);
    return v;
}

// TDM 2-D tile load: rows x 256 halves, row stride 256 halves in memory,
// LDS rows padded 256 -> 264 halves (+16B per 512B) via D# pad fields.
// D# layout per CDNA5 ISA 8.3-8.5. data_size=1 (2 bytes). type=2.
__device__ __forceinline__ void tdm_load_rows(const _Float16* gsrc, void* ldst, int rows) {
    unsigned long long ga = (unsigned long long)(uintptr_t)gsrc;
    unsigned lo = (unsigned)(uintptr_t)ldst;   // generic LDS ptr: low 32 bits = LDS byte offset
    u32x4 g0 = { 1u,                           // count=1 (valid user descriptor)
                 lo,                           // lds_addr
                 (unsigned)ga,                 // global_addr[31:0]
                 (unsigned)((ga >> 32) & 0x01FFFFFFu) | 0x80000000u }; // ga[56:32] | type=2
    i32x8 g1;
    g1[0] = (1 << 16)                          // data_size = 1 -> 2 bytes
          | (1 << 20)                          // pad_enable
          | (6 << 22)                          // pad_interval: 128 dwords (512B)
          | (3 << 25);                         // pad_amount: 4 dwords (16B)
    g1[1] = (int)(256u << 16);                 // tensor_dim0[15:0]=256 in bits 63:48
    g1[2] = (int)((unsigned)rows << 16);       // tensor_dim0 hi=0 | tensor_dim1[15:0]=rows
    g1[3] = (int)(256u << 16);                 // tensor_dim1 hi=0 | tile_dim0=256
    g1[4] = rows;                              // tile_dim1=rows | tile_dim2=0
    g1[5] = 256;                               // tensor_dim0_stride[31:0] = 256 elements
    g1[6] = 0;
    g1[7] = 0;
    i32x4 z4 = { 0, 0, 0, 0 };
    i32x8 z8 = { 0, 0, 0, 0, 0, 0, 0, 0 };
    __builtin_amdgcn_tensor_load_to_lds(g0, g1, z4, z4, z8, 0);
}

// ---------------- Kernel 0: weight f32 -> f16 conversion ----------------
__global__ void cvt_w_kernel(const float* __restrict__ qkvw, const float* __restrict__ outw,
                             _Float16* __restrict__ qkvw16, _Float16* __restrict__ outw16) {
    int i = blockIdx.x * 256 + threadIdx.x;
    if (i < 3 * C_ * C_) qkvw16[i] = (_Float16)qkvw[i];
    if (i < C_ * C_)     outw16[i] = (_Float16)outw[i];
}

// ---------------- Kernel 1: GroupNorm -> f16  h[b][c][l] ----------------
__global__ __launch_bounds__(256) void gn_kernel(const float* __restrict__ x,
                                                 const float* __restrict__ gw,
                                                 const float* __restrict__ gb,
                                                 _Float16* __restrict__ h16) {
    __shared__ float red[17];
    int bg = blockIdx.x;
    int b = bg / G_, g = bg % G_;
    int c0 = g * CPG;
    const float* xp = x + ((size_t)b * C_ + c0) * L_;
    const int N = CPG * L_;                       // 16384
    float s = 0.f, s2 = 0.f;
    for (int i = threadIdx.x; i < N; i += 256) {
        float v = xp[i];
        s += v; s2 += v * v;
    }
    for (int m = 1; m <= 16; m <<= 1) { s += __shfl_xor(s, m); s2 += __shfl_xor(s2, m); }
    int wave = threadIdx.x >> 5, lane = threadIdx.x & 31;
    if (lane == 0) { red[wave] = s; red[8 + wave] = s2; }
    __syncthreads();
    if (threadIdx.x == 0) {
        float ts = 0.f, ts2 = 0.f;
        for (int w = 0; w < 8; ++w) { ts += red[w]; ts2 += red[8 + w]; }
        float mean = ts * (1.0f / N);
        float var  = ts2 * (1.0f / N) - mean * mean;
        red[0] = mean;
        red[1] = rsqrtf(var + 1e-5f);
    }
    __syncthreads();
    float mean = red[0], inv = red[1];
    _Float16* hp = h16 + ((size_t)b * C_ + c0) * L_;
    for (int i = threadIdx.x; i < N; i += 256) {
        int c = c0 + (i >> 11);
        float v = (xp[i] - mean) * inv * gw[c] + gb[c];
        hp[i] = (_Float16)v;
    }
}

// ---------------- Kernel 2: QKV GEMM + layout scatter -------------------
// qkv[o][l] = sum_c W[o][c] h[c][l] + bias[o], per batch.
// q (scaled by log2e/sqrt(D)) -> qT[b][h][l][d], k -> kT[b][h][l][d], v -> vN[b][h][d][l]
__global__ __launch_bounds__(256) void qkv_kernel(const _Float16* __restrict__ w16,
                                                  const float* __restrict__ bias,
                                                  const _Float16* __restrict__ h16,
                                                  _Float16* __restrict__ qT,
                                                  _Float16* __restrict__ kT,
                                                  _Float16* __restrict__ vN) {
    __shared__ _Float16 lA[64 * AST];   // W rows [o][c], row-major, padded (TDM)
    __shared__ _Float16 lB[32 * BST];   // h^T   [l][c], row-major, padded (manual transpose)
    int b  = blockIdx.z;
    int o0 = blockIdx.y * 64;
    int l0 = blockIdx.x * 32;
    int t = threadIdx.x;
    int wave = t >> 5, lane = t & 31;

    if (wave == 0)  // async DMA of the A tile while everyone transposes B
        tdm_load_rows(w16 + (size_t)o0 * C_, lA, 64);

    { // stage Bt: transpose h[b][c][l0..l0+31] -> lB[l][c]
        const _Float16* hb = h16 + (size_t)b * C_ * L_;
        for (int i = t; i < C_ * 32; i += 256) {
            int c = i >> 5, l = i & 31;
            lB[l * BST + c] = hb[(size_t)c * L_ + l0 + l];
        }
    }
    if (wave == 0) __builtin_amdgcn_s_wait_tensorcnt(0);
    __syncthreads();

    int wm = wave >> 1, wn = wave & 1;          // 4 M-subtiles x 2 N-subtiles
    int lan15 = lane & 15, khi = lane >> 4;
    const _Float16* ap = &lA[(wm * 16 + lan15) * AST];
    const _Float16* bp = &lB[(wn * 16 + lan15) * BST];

    v8f acc = {};
#pragma unroll
    for (int kk = 0; kk < C_; kk += 32) {
        v16h a  = ldfrag(ap + kk + khi * 8,   ap + kk + 16 + khi * 8);
        v16h bb = ldfrag(bp + kk + khi * 8,   bp + kk + 16 + khi * 8);
        acc = __builtin_amdgcn_wmma_f32_16x16x32_f16(false, a, false, bb, (short)0, acc, false, false);
    }

    // scale folds 1/sqrt(D) AND log2(e) so attention uses raw v_exp (exp2)
    const float QSCALE = 0.17677669529663687f * 1.4426950408889634f;
    int l = l0 + wn * 16 + lan15;
    int sec = (o0 >> 8);                         // 0=q, 1=k, 2=v (uniform per block)
#pragma unroll
    for (int j = 0; j < 8; ++j) {
        int o = o0 + wm * 16 + j + 8 * khi;
        float v = acc[j] + bias[o];
        int c = o & (C_ - 1);
        int hh = c >> 5, d = c & 31;
        if (sec == 0)
            qT[(((size_t)b * H_ + hh) * L_ + l) * D_ + d] = (_Float16)(v * QSCALE);
        else if (sec == 1)
            kT[(((size_t)b * H_ + hh) * L_ + l) * D_ + d] = (_Float16)v;
        else
            vN[(((size_t)b * H_ + hh) * D_ + d) * L_ + l] = (_Float16)v;
    }
}

// ---------------- Kernel 3: flash attention per (b,h) -------------------
// S = qT kT^T in log2 units, online softmax via exp2, O = P V^T.
// Output attnT[b][l][c] (c = h*32+d) as f16.
__global__ __launch_bounds__(128) void attn_kernel(const _Float16* __restrict__ qT,
                                                   const _Float16* __restrict__ kT,
                                                   const _Float16* __restrict__ vN,
                                                   _Float16* __restrict__ attnT) {
    __shared__ _Float16 lP[4][16 * PST];
    int bh = blockIdx.y;
    int b = bh >> 3, hh = bh & 7;
    int wave = threadIdx.x >> 5, lane = threadIdx.x & 31;
    int lan15 = lane & 15, khi = lane >> 4;
    int l0 = blockIdx.x * 64 + wave * 16;

    const _Float16* qp = qT + (size_t)bh * L_ * D_;
    const _Float16* kp = kT + (size_t)bh * L_ * D_;
    const _Float16* vp = vN + (size_t)bh * D_ * L_;

    const _Float16* qrow = qp + (size_t)(l0 + lan15) * D_;
    v16h qa = ldfrag(qrow + khi * 8, qrow + 16 + khi * 8);

    float rmax[8], rsum[8];
#pragma unroll
    for (int j = 0; j < 8; ++j) { rmax[j] = -3.402823466e+38f; rsum[j] = 0.f; }
    v8f oa0 = {}, oa1 = {};
    _Float16* myP = &lP[wave][0];

    for (int it = 0; it < L_ / 32; ++it) {
        int m0 = it * 32;
        const _Float16* kr = kp + (size_t)(m0 + lan15) * D_ + khi * 8;
        v16h kb0 = ldfrag(kr, kr + 16);
        kr += 16 * D_;
        v16h kb1 = ldfrag(kr, kr + 16);
        const _Float16* vr = vp + (size_t)lan15 * L_ + m0 + khi * 8;
        v16h vb0 = ldfrag(vr, vr + 16);
        vr += 16 * L_;
        v16h vb1 = ldfrag(vr, vr + 16);

        if (it + 1 < L_ / 32) {
            __builtin_prefetch(kp + (size_t)(m0 + 32 + lan15) * D_, 0, 0);
            __builtin_prefetch(vp + (size_t)lan15 * L_ + m0 + 32, 0, 0);
        }

        v8f zero = {};
        v8f s0 = __builtin_amdgcn_wmma_f32_16x16x32_f16(false, qa, false, kb0, (short)0, zero, false, false);
        v8f s1 = __builtin_amdgcn_wmma_f32_16x16x32_f16(false, qa, false, kb1, (short)0, zero, false, false);

        // online softmax in log2 domain; row M = j + 8*khi lives in one 16-lane half
#pragma unroll
        for (int j = 0; j < 8; ++j) {
            float tmax = redmax16(fmaxf(s0[j], s1[j]));
            float mnew = fmaxf(rmax[j], tmax);
            float alpha = __builtin_amdgcn_exp2f(rmax[j] - mnew);
            float p0 = __builtin_amdgcn_exp2f(s0[j] - mnew);
            float p1 = __builtin_amdgcn_exp2f(s1[j] - mnew);
            rsum[j] = rsum[j] * alpha + redsum16(p0 + p1);
            rmax[j] = mnew;
            oa0[j] *= alpha;
            oa1[j] *= alpha;
            int row = j + 8 * khi;
            myP[row * PST + lan15]      = (_Float16)p0;
            myP[row * PST + 16 + lan15] = (_Float16)p1;
        }
        // P fragment (A layout) back from per-wave LDS (in-order within wave)
        const _Float16* pr = myP + lan15 * PST;
        v16h pa = ldfrag(pr + khi * 8, pr + 16 + khi * 8);

        oa0 = __builtin_amdgcn_wmma_f32_16x16x32_f16(false, pa, false, vb0, (short)0, oa0, false, false);
        oa1 = __builtin_amdgcn_wmma_f32_16x16x32_f16(false, pa, false, vb1, (short)0, oa1, false, false);
    }

    _Float16* op = attnT + (size_t)b * L_ * C_;
#pragma unroll
    for (int j = 0; j < 8; ++j) {
        int l = l0 + j + 8 * khi;
        float inv = 1.0f / rsum[j];
        op[(size_t)l * C_ + hh * 32 + lan15]      = (_Float16)(oa0[j] * inv);
        op[(size_t)l * C_ + hh * 32 + 16 + lan15] = (_Float16)(oa1[j] * inv);
    }
}

// ---------------- Kernel 4: out projection + bias + residual ------------
__global__ __launch_bounds__(256) void outp_kernel(const _Float16* __restrict__ w16,
                                                   const float* __restrict__ bias,
                                                   const _Float16* __restrict__ attnT,
                                                   const float* __restrict__ x,
                                                   float* __restrict__ out) {
    __shared__ _Float16 lA[64 * AST];
    __shared__ _Float16 lB[32 * BST];
    int b  = blockIdx.z;
    int o0 = blockIdx.y * 64;
    int l0 = blockIdx.x * 32;
    int t = threadIdx.x;
    int wave = t >> 5, lane = t & 31;

    // both stage-ins are plain strided 2-D copies -> Tensor Data Mover
    if (wave == 0)
        tdm_load_rows(w16 + (size_t)o0 * C_, lA, 64);
    else if (wave == 1)
        tdm_load_rows(attnT + ((size_t)b * L_ + l0) * C_, lB, 32);
    if (wave < 2) __builtin_amdgcn_s_wait_tensorcnt(0);
    __syncthreads();

    int wm = wave >> 1, wn = wave & 1;
    int lan15 = lane & 15, khi = lane >> 4;
    const _Float16* ap = &lA[(wm * 16 + lan15) * AST];
    const _Float16* bp = &lB[(wn * 16 + lan15) * BST];

    v8f acc = {};
#pragma unroll
    for (int kk = 0; kk < C_; kk += 32) {
        v16h a  = ldfrag(ap + kk + khi * 8,   ap + kk + 16 + khi * 8);
        v16h bb = ldfrag(bp + kk + khi * 8,   bp + kk + 16 + khi * 8);
        acc = __builtin_amdgcn_wmma_f32_16x16x32_f16(false, a, false, bb, (short)0, acc, false, false);
    }

    int l = l0 + wn * 16 + lan15;
#pragma unroll
    for (int j = 0; j < 8; ++j) {
        int o = o0 + wm * 16 + j + 8 * khi;
        size_t idx = ((size_t)b * C_ + o) * L_ + l;
        out[idx] = acc[j] + bias[o] + x[idx];
    }
}

extern "C" void kernel_launch(void* const* d_in, const int* in_sizes, int n_in,
                              void* d_out, int out_size, void* d_ws, size_t ws_size,
                              hipStream_t stream) {
    const float* x     = (const float*)d_in[0];
    const float* gn_w  = (const float*)d_in[1];
    const float* gn_b  = (const float*)d_in[2];
    const float* qkv_w = (const float*)d_in[3];
    const float* qkv_b = (const float*)d_in[4];
    const float* out_w = (const float*)d_in[5];
    const float* out_b = (const float*)d_in[6];
    float* out = (float*)d_out;

    char* ws = (char*)d_ws;
    const size_t MB = 1ull << 20;
    _Float16* h16   = (_Float16*)(ws + 0 * MB);    // 4 MB  [b][c][l]
    _Float16* qT    = (_Float16*)(ws + 4 * MB);    // 4 MB  [b][h][l][d] (scaled)
    _Float16* kT    = (_Float16*)(ws + 8 * MB);    // 4 MB  [b][h][l][d]
    _Float16* vN    = (_Float16*)(ws + 12 * MB);   // 4 MB  [b][h][d][l]
    _Float16* attnT = (_Float16*)(ws + 16 * MB);   // 4 MB  [b][l][c]
    _Float16* w16q  = (_Float16*)(ws + 20 * MB);   // 384 KB
    _Float16* w16o  = (_Float16*)(ws + 20 * MB + 512 * 1024); // 128 KB

    cvt_w_kernel<<<(3 * C_ * C_ + 255) / 256, 256, 0, stream>>>(qkv_w, out_w, w16q, w16o);
    gn_kernel<<<B_ * G_, 256, 0, stream>>>(x, gn_w, gn_b, h16);
    dim3 g2(L_ / 32, (3 * C_) / 64, B_);
    qkv_kernel<<<g2, 256, 0, stream>>>(w16q, qkv_b, h16, qT, kT, vN);
    dim3 g3(L_ / 64, B_ * H_);
    attn_kernel<<<g3, 128, 0, stream>>>(qT, kT, vN, attnT);
    dim3 g4(L_ / 32, C_ / 64, B_);
    outp_kernel<<<g4, 256, 0, stream>>>(w16o, out_b, attnT, x, out);
}
